// HofmanSwarm_15015205666876
// MI455X (gfx1250) — matrix operand order repaired and checked by reading, not compiled, not run on hardware
//
#include <hip/hip_runtime.h>
#include <hip/hip_bf16.h>

typedef __attribute__((ext_vector_type(16))) __bf16 v16bf;
typedef __attribute__((ext_vector_type(8)))  float  v8f;

#define D_DIM 256
#define A_DIM 64
#define NS    8
#define T_LEN 128
#define V_DIM 32000
#define H_DIM 512   // 2*D

__device__ __forceinline__ v8f wmma_bf16(v16bf a, v16bf b, v8f c) {
    return __builtin_amdgcn_wmma_f32_16x16x32_bf16(
        false, a, false, b, (short)0, c, false, false);
}

// ---- optional gfx1250 async global->LDS copy (ASYNCcnt path) ----
#if defined(__HIP_DEVICE_COMPILE__) && __has_builtin(__builtin_amdgcn_global_load_async_to_lds_b128)
#define HAVE_ASYNC_CP 1
typedef int v4i_async __attribute__((__vector_size__(16)));
typedef __attribute__((address_space(1))) v4i_async* async_gptr;
typedef __attribute__((address_space(3))) v4i_async* async_lptr;
__device__ __forceinline__ void async_cp16(const __bf16* g, __bf16* l) {
    __builtin_amdgcn_global_load_async_to_lds_b128(
        (async_gptr)(uintptr_t)g, (async_lptr)(uintptr_t)l, 0, 0);
}
__device__ __forceinline__ void async_wait0() {
#if __has_builtin(__builtin_amdgcn_s_wait_asynccnt)
    __builtin_amdgcn_s_wait_asynccnt(0);
#else
    asm volatile("s_wait_asynccnt 0" ::: "memory");
#endif
}
#endif

// ---- vectorized fragment builders (two 16B loads per fragment) ----
__device__ __forceinline__ v16bf ld_frag_2x16B(const __bf16* p0, const __bf16* p1) {
    union { v16bf v; uint4 q[2]; } u;
    u.q[0] = *(const uint4*)p0;
    u.q[1] = *(const uint4*)p1;
    return u.v;
}

// A: 16x32 bf16 tile from row-major [M][K]. lanes0-15: K 0..15 ; lanes16-31: K 16..31
__device__ __forceinline__ v16bf frag_a_bf16(const __bf16* base, int ld, int m0, int k0, int lane) {
    int m  = m0 + (lane & 15);
    int kb = k0 + ((lane < 16) ? 0 : 16);
    const __bf16* row = base + (size_t)m * ld + kb;
    return ld_frag_2x16B(row, row + 8);
}

// A from f32 LDS (convert to bf16)
__device__ __forceinline__ v16bf frag_a_f32(const float* base, int ld, int m0, int k0, int lane) {
    int m  = m0 + (lane & 15);
    int kb = k0 + ((lane < 16) ? 0 : 16);
    const float* p = base + (size_t)m * ld;
    v16bf a;
#pragma unroll
    for (int e = 0; e < 8; ++e) a[e]     = (__bf16)p[kb + e];
#pragma unroll
    for (int e = 0; e < 8; ++e) a[8 + e] = (__bf16)p[kb + 8 + e];
    return a;
}

// B: 32x16 tile; source is TRANSPOSED weight [N][K] row-major (per-lane K contiguous).
// lanes0-15: K 0..15 ; lanes16-31: K 16..31 ; N = lane&15
__device__ __forceinline__ v16bf frag_b_bf16T(const __bf16* base, int ld, int k0, int n0, int lane) {
    int n  = n0 + (lane & 15);
    int kb = k0 + ((lane < 16) ? 0 : 16);
    const __bf16* p = base + (size_t)n * ld + kb;
    return ld_frag_2x16B(p, p + 8);
}

// B from f32 LDS stored natural [K][N] row-major (column gather + convert)
__device__ __forceinline__ v16bf frag_b_f32_KN(const float* base, int ld, int k0, int n0, int lane) {
    int n  = n0 + (lane & 15);
    int kb = k0 + ((lane < 16) ? 0 : 16);
    v16bf b;
#pragma unroll
    for (int e = 0; e < 16; ++e) b[e] = (__bf16)base[(size_t)(kb + e) * ld + n];
    return b;
}

__device__ __forceinline__ float gelu_exact(float x) {
    return 0.5f * x * (1.0f + erff(x * 0.70710678118654752f));
}

// ---------------- conversion kernels ----------------
__global__ void cvt_bf16(const float* __restrict__ src, __bf16* __restrict__ dst, int n) {
    int i = blockIdx.x * 256 + threadIdx.x;
    if (i < n) dst[i] = (__bf16)src[i];
}

// src [rows][cols] f32 -> dst [cols][rows] bf16
__global__ void cvt_bf16_T(const float* __restrict__ src, __bf16* __restrict__ dst, int rows, int cols) {
    int i = blockIdx.x * 256 + threadIdx.x;
    if (i < rows * cols) {
        int r = i / cols, c = i % cols;
        dst[(size_t)c * rows + r] = (__bf16)src[i];
    }
}

// ---------------- recurrence kernel: one block per batch element ----------------
__global__ __launch_bounds__(256) void swarm_recurrence(
    const float* __restrict__ perception, const float* __restrict__ pos,
    const int*   __restrict__ idx,
    const __bf16* __restrict__ ANbf, const __bf16* __restrict__ Wsbf,
    const float* __restrict__ bs,
    const __bf16* __restrict__ WgT,  const float* __restrict__ bg,
    const __bf16* __restrict__ WcT,  const float* __restrict__ bc,
    const float* __restrict__ ln_g,  const float* __restrict__ ln_b,
    const float* __restrict__ agent_imp,
    const __bf16* __restrict__ W1bf, const float* __restrict__ b1,
    __bf16* __restrict__ H)
{
    constexpr int SLD = 257;   // f32 row stride (bank-conflict-free column walks)
    constexpr int ILD = 272;   // bf16 row stride (row pitch 544B, 16B aligned)

    extern __shared__ char smem[];
    float*  S    = (float*)smem;                       // [64][257] f32  states
    float*  G    = S + A_DIM * SLD;                    // [64][257] f32  gate pre-act
    __bf16* I    = (__bf16*)(G + A_DIM * SLD);         // [64][272] bf16 interaction
    float*  pbuf = (float*)(I + A_DIM * ILD);          // [256]
    float*  iw   = pbuf + D_DIM;                       // [64]
    float*  cons = iw + A_DIM;                         // [256]

    const int b    = blockIdx.x;
    const int tid  = threadIdx.x;
    const int wave = tid >> 5;
    const int lane = tid & 31;

    for (int i = tid; i < A_DIM * SLD; i += 256) S[i] = 0.0f;
    if (tid < A_DIM) iw[tid] = agent_imp[tid];
    __syncthreads();
    if (tid == 0) {  // softmax over 64 agent-importance values (once)
        float mx = iw[0];
        for (int i = 1; i < A_DIM; ++i) mx = fmaxf(mx, iw[i]);
        float s = 0.0f;
        for (int i = 0; i < A_DIM; ++i) { iw[i] = expf(iw[i] - mx); s += iw[i]; }
        float inv = 1.0f / s;
        for (int i = 0; i < A_DIM; ++i) iw[i] *= inv;
    }
    __syncthreads();

    for (int t = 0; t < T_LEN; ++t) {
        // --- percept p_t = perception[idx] + pos[t] ---
        int token = idx[b * T_LEN + t];
        pbuf[tid] = perception[(size_t)token * D_DIM + tid] + pos[t * D_DIM + tid];
        __syncthreads();

        // --- sensory = p @ Ws + bs ; states[:8,:] += sensory ---
        {
            float acc[NS];
#pragma unroll
            for (int u = 0; u < NS; ++u) acc[u] = bs[u * D_DIM + tid];
            for (int k = 0; k < D_DIM; ++k) {
                float pk = pbuf[k];
                const __bf16* wr = Wsbf + (size_t)k * (NS * D_DIM) + tid;
#pragma unroll
                for (int u = 0; u < NS; ++u) acc[u] += pk * (float)wr[u * D_DIM];
            }
#pragma unroll
            for (int u = 0; u < NS; ++u) S[u * SLD + tid] += acc[u];
        }
        __syncthreads();

        // --- interaction I = AN @ S   (M=64,K=64,N=256) -> bf16 LDS ---
        for (int tile = wave; tile < 64; tile += 8) {
            int m0 = (tile >> 4) << 4, n0 = (tile & 15) << 4;
            v8f c = {};
#pragma unroll
            for (int k0 = 0; k0 < A_DIM; k0 += 32) {
                v16bf a  = frag_a_bf16(ANbf, A_DIM, m0, k0, lane);
                v16bf bb = frag_b_f32_KN(S, SLD, k0, n0, lane);
                c = wmma_bf16(a, bb, c);
            }
            int col   = n0 + (lane & 15);
            int rbase = m0 + ((lane >> 4) << 3);
#pragma unroll
            for (int r = 0; r < 8; ++r) I[(rbase + r) * ILD + col] = (__bf16)c[r];
        }
        __syncthreads();

        // --- gate preact G = S@Wg[:256] + I@Wg[256:] + bg  (K=512) ---
        for (int tile = wave; tile < 64; tile += 8) {
            int m0 = (tile >> 4) << 4, n0 = (tile & 15) << 4;
            v8f c = {};
#pragma unroll
            for (int k0 = 0; k0 < D_DIM; k0 += 32) {
                v16bf a  = frag_a_f32(S, SLD, m0, k0, lane);
                v16bf bb = frag_b_bf16T(WgT, H_DIM, k0, n0, lane);
                c = wmma_bf16(a, bb, c);
            }
#pragma unroll
            for (int k0 = 0; k0 < D_DIM; k0 += 32) {
                v16bf a  = frag_a_bf16(I, ILD, m0, k0, lane);
                v16bf bb = frag_b_bf16T(WgT, H_DIM, D_DIM + k0, n0, lane);
                c = wmma_bf16(a, bb, c);
            }
            int col   = n0 + (lane & 15);
            int rbase = m0 + ((lane >> 4) << 3);
            float bgc = bg[col];
#pragma unroll
            for (int r = 0; r < 8; ++r) G[(rbase + r) * SLD + col] = c[r] + bgc;
        }
        __syncthreads();

        // --- candidate = gelu(I @ Wc + bc), fused gated state update into S ---
        for (int tile = wave; tile < 64; tile += 8) {
            int m0 = (tile >> 4) << 4, n0 = (tile & 15) << 4;
            v8f c = {};
#pragma unroll
            for (int k0 = 0; k0 < D_DIM; k0 += 32) {
                v16bf a  = frag_a_bf16(I, ILD, m0, k0, lane);
                v16bf bb = frag_b_bf16T(WcT, D_DIM, k0, n0, lane);
                c = wmma_bf16(a, bb, c);
            }
            int col   = n0 + (lane & 15);
            int rbase = m0 + ((lane >> 4) << 3);
            float bcc = bc[col];
#pragma unroll
            for (int r = 0; r < 8; ++r) {
                int row   = rbase + r;
                float cc  = c[r] + bcc;
                float ge  = gelu_exact(cc);
                float gp  = G[row * SLD + col];
                float sg  = 1.0f / (1.0f + __expf(-gp));
                float so  = S[row * SLD + col];
                S[row * SLD + col] = sg * ge + (1.0f - sg) * so;
            }
        }
        __syncthreads();

        // --- LayerNorm each row of S (4 lanes per row) ---
        {
            int row = tid >> 2, part = tid & 3;
            float* sr = S + row * SLD + part * 64;
            float sum = 0.0f, sq = 0.0f;
#pragma unroll 8
            for (int j = 0; j < 64; ++j) { float x = sr[j]; sum += x; sq += x * x; }
            sum += __shfl_xor(sum, 1, 32);  sq += __shfl_xor(sq, 1, 32);
            sum += __shfl_xor(sum, 2, 32);  sq += __shfl_xor(sq, 2, 32);
            float m   = sum * (1.0f / 256.0f);
            float var = sq * (1.0f / 256.0f) - m * m;
            float inv = rsqrtf(var + 1e-5f);
#pragma unroll 8
            for (int j = 0; j < 64; ++j) {
                int d = part * 64 + j;
                float x = sr[j];
                sr[j] = (x - m) * inv * ln_g[d] + ln_b[d];
            }
        }
        __syncthreads();

        // --- consensus + h = gelu(consensus @ W1 + b1) -> workspace (bf16) ---
        {
            float cs = 0.0f;
#pragma unroll 8
            for (int a = 0; a < A_DIM; ++a) cs += iw[a] * S[a * SLD + tid];
            cons[tid] = cs;
        }
        __syncthreads();
        {
            float h0 = b1[tid], h1 = b1[tid + 256];
            for (int k = 0; k < D_DIM; ++k) {
                float ck = cons[k];
                const __bf16* wr = W1bf + (size_t)k * H_DIM;
                h0 += ck * (float)wr[tid];
                h1 += ck * (float)wr[tid + 256];
            }
            __bf16* hrow = H + (size_t)(b * T_LEN + t) * H_DIM;
            hrow[tid]       = (__bf16)gelu_exact(h0);
            hrow[tid + 256] = (__bf16)gelu_exact(h1);
        }
        __syncthreads();
    }
}

// ---------------- logits GEMM: [M=B*T,512] x [512,32000] + b2 ----------------
// block tile 128x128, K staged 64 wide (async global->LDS when available),
// 8 waves (2M x 4N), each wave 4 Mtiles x 2 Ntiles x 2 Ksteps per stage.
__global__ __launch_bounds__(256) void logits_gemm(
    const __bf16* __restrict__ H,    // [M][512]
    const __bf16* __restrict__ W2T,  // [32000][512]
    const float*  __restrict__ b2,
    float* __restrict__ out)         // [M][32000]
{
    __shared__ __attribute__((aligned(16))) __bf16 As[128][64];   // 16KB
    __shared__ __attribute__((aligned(16))) __bf16 Bs[128][64];   // 16KB

    const int n0b  = blockIdx.x * 128;
    const int m0b  = blockIdx.y * 128;
    const int tid  = threadIdx.x;
    const int wave = tid >> 5;
    const int lane = tid & 31;
    const int wm   = wave >> 2;      // 0..1  -> M offset wm*64
    const int wn   = wave & 3;       // 0..3  -> N offset wn*32

    v8f acc[4][2];
#pragma unroll
    for (int i = 0; i < 4; ++i)
#pragma unroll
        for (int j = 0; j < 2; ++j) acc[i][j] = (v8f){};

    const int r = tid >> 1, h = tid & 1;   // staging: row 0..127, 32-elem half

    for (int kk = 0; kk < H_DIM; kk += 64) {
        const __bf16* asrc = H   + (size_t)(m0b + r) * H_DIM + kk + h * 32;
        const __bf16* bsrc = W2T + (size_t)(n0b + r) * H_DIM + kk + h * 32;
        __bf16* adst = &As[r][h * 32];
        __bf16* bdst = &Bs[r][h * 32];
        if (kk + 64 < H_DIM) {   // prefetch next K slab (global_prefetch_b8)
            __builtin_prefetch(asrc + 64, 0, 0);
            __builtin_prefetch(bsrc + 64, 0, 0);
        }
#ifdef HAVE_ASYNC_CP
#pragma unroll
        for (int c = 0; c < 4; ++c) {
            async_cp16(asrc + c * 8, adst + c * 8);
            async_cp16(bsrc + c * 8, bdst + c * 8);
        }
        async_wait0();
#else
#pragma unroll
        for (int c = 0; c < 4; ++c) {
            ((uint4*)adst)[c] = ((const uint4*)asrc)[c];
            ((uint4*)bdst)[c] = ((const uint4*)bsrc)[c];
        }
#endif
        __syncthreads();

#pragma unroll
        for (int ks = 0; ks < 2; ++ks) {
            v16bf bfr[2];
#pragma unroll
            for (int nt = 0; nt < 2; ++nt)
                bfr[nt] = frag_b_bf16T(&Bs[0][0], 64, ks * 32, wn * 32 + nt * 16, lane);
#pragma unroll
            for (int mt = 0; mt < 4; ++mt) {
                v16bf afr = frag_a_bf16(&As[0][0], 64, wm * 64 + mt * 16, ks * 32, lane);
#pragma unroll
                for (int nt = 0; nt < 2; ++nt)
                    acc[mt][nt] = wmma_bf16(afr, bfr[nt], acc[mt][nt]);
            }
        }
        __syncthreads();
    }

    // epilogue: add bias, write f32 logits
#pragma unroll
    for (int mt = 0; mt < 4; ++mt) {
        int rbase = m0b + wm * 64 + mt * 16 + ((lane >> 4) << 3);
#pragma unroll
        for (int nt = 0; nt < 2; ++nt) {
            int col = n0b + wn * 32 + nt * 16 + (lane & 15);
            float bias = b2[col];
#pragma unroll
            for (int rr = 0; rr < 8; ++rr)
                out[(size_t)(rbase + rr) * V_DIM + col] = acc[mt][nt][rr] + bias;
        }
    }
}

extern "C" void kernel_launch(void* const* d_in, const int* in_sizes, int n_in,
                              void* d_out, int out_size, void* d_ws, size_t ws_size,
                              hipStream_t stream) {
    const float* perception = (const float*)d_in[0];
    const float* pos        = (const float*)d_in[1];
    const float* Ws         = (const float*)d_in[2];
    const float* bs         = (const float*)d_in[3];
    const float* AN         = (const float*)d_in[4];
    const float* Wg         = (const float*)d_in[5];
    const float* bg         = (const float*)d_in[6];
    const float* Wc         = (const float*)d_in[7];
    const float* bc         = (const float*)d_in[8];
    const float* ln_g       = (const float*)d_in[9];
    const float* ln_b       = (const float*)d_in[10];
    const float* ai         = (const float*)d_in[11];
    const float* W1         = (const float*)d_in[12];
    const float* b1         = (const float*)d_in[13];
    const float* W2         = (const float*)d_in[14];
    const float* b2         = (const float*)d_in[15];
    const int*   idx        = (const int*)d_in[16];

    const int B = in_sizes[16] / T_LEN;   // 16
    const int M = B * T_LEN;              // 2048

    char* ws = (char*)d_ws;
    size_t off = 0;
    auto take = [&](size_t bytes) { char* p = ws + off; off = (off + bytes + 255) & ~(size_t)255; return p; };
    __bf16* ANbf = (__bf16*)take((size_t)A_DIM * A_DIM * 2);
    __bf16* Wsbf = (__bf16*)take((size_t)D_DIM * NS * D_DIM * 2);
    __bf16* WcT  = (__bf16*)take((size_t)D_DIM * D_DIM * 2);
    __bf16* WgT  = (__bf16*)take((size_t)H_DIM * D_DIM * 2);
    __bf16* W1bf = (__bf16*)take((size_t)D_DIM * H_DIM * 2);
    __bf16* W2T  = (__bf16*)take((size_t)H_DIM * V_DIM * 2);
    __bf16* H    = (__bf16*)take((size_t)M * H_DIM * 2);
    (void)ws_size; (void)n_in; (void)out_size;

    // phase 1: weight conversion / transposition (bf16 for WMMA)
    cvt_bf16  <<<(A_DIM * A_DIM + 255) / 256, 256, 0, stream>>>(AN, ANbf, A_DIM * A_DIM);
    cvt_bf16  <<<(D_DIM * NS * D_DIM + 255) / 256, 256, 0, stream>>>(Ws, Wsbf, D_DIM * NS * D_DIM);
    cvt_bf16_T<<<(D_DIM * D_DIM + 255) / 256, 256, 0, stream>>>(Wc, WcT, D_DIM, D_DIM);
    cvt_bf16_T<<<(H_DIM * D_DIM + 255) / 256, 256, 0, stream>>>(Wg, WgT, H_DIM, D_DIM);
    cvt_bf16  <<<(D_DIM * H_DIM + 255) / 256, 256, 0, stream>>>(W1, W1bf, D_DIM * H_DIM);
    cvt_bf16_T<<<(H_DIM * V_DIM + 255) / 256, 256, 0, stream>>>(W2, W2T, H_DIM, V_DIM);

    // phase 2: recurrence, one workgroup per batch element, state in LDS
    constexpr int SLD = 257, ILD = 272;
    size_t smem = (size_t)A_DIM * SLD * 4   // S
                + (size_t)A_DIM * SLD * 4   // G
                + (size_t)A_DIM * ILD * 2   // I
                + (D_DIM + A_DIM + D_DIM) * 4;
    swarm_recurrence<<<B, 256, smem, stream>>>(
        perception, pos, idx, ANbf, Wsbf, bs, WgT, bg, WcT, bc,
        ln_g, ln_b, ai, W1bf, b1, H);

    // phase 3: logits = H @ W2 + b2   (M x 32000, K=512)
    dim3 grid(V_DIM / 128, M / 128);
    logits_gemm<<<grid, 256, 0, stream>>>(H, W2T, b2, (float*)d_out);
}
